// BinaryMemoryTree_90812788506711
// MI455X (gfx1250) — matrix operand order repaired and checked by reading, not compiled, not run on hardware
//
#include <hip/hip_runtime.h>

// Problem constants (match reference)
#define B_   8
#define LK   32768
#define D_   128
#define NC   2

#define ROWS_PER_WG    512
#define WAVES_PER_WG   8
#define TILES_PER_WAVE 4   // 4 * 8 waves * 16 rows = 512 rows per WG

typedef __attribute__((ext_vector_type(16))) __bf16 v16bf;
typedef __attribute__((ext_vector_type(8)))  float  v8f;

// LDS B-fragment storage: [c][nt][kb][lane] -> 16 bf16 (32 B) per lane = 64 KB
#define NFRAG (NC * 8 * 4 * 32)

__global__ __launch_bounds__(256)
void bmt_wmma_kernel(const float* __restrict__ q,
                     const float* __restrict__ v,
                     const float* __restrict__ mem,
                     float* __restrict__ out)
{
    __shared__ v16bf Bsm[NFRAG];   // 64 KB

    const int tid  = threadIdx.x;
    const int lane = tid & 31;
    const int wave = tid >> 5;
    const int b       = blockIdx.x >> 6;            // 64 row-blocks per batch
    const int rowBase = (blockIdx.x & 63) * ROWS_PER_WG;

    // ---------------- Phase 1: pack memory[b][c] -> bf16 B fragments in LDS
    // Fragment F = ((c*8 + nt)*4 + kb)*32 + L
    // Lane L holds column n = nt*16 + (L&15), K values k = kb*32 + (L>>4)*16 + i
    // (contiguous k order, same order as the A fragments)
    for (int F = tid; F < NFRAG; F += 256) {
        const int L  = F & 31;
        const int kb = (F >> 5) & 3;
        const int nt = (F >> 7) & 7;
        const int c  = F >> 10;
        const int n  = nt * 16 + (L & 15);
        const int k0 = kb * 32 + ((L >> 4) << 4);
        const float* Ms = mem + ((size_t)(b * NC + c)) * D_ * D_;   // [d][e] row-major
        v16bf frag;
        #pragma unroll
        for (int i = 0; i < 16; ++i)
            frag[i] = (__bf16)Ms[(size_t)(k0 + i) * D_ + n];
        Bsm[F] = frag;
    }
    __syncthreads();

    const int hi = lane >> 4;    // wave half
    const int m  = lane & 15;    // row within tile (A), column (C/D)

    for (int t = 0; t < TILES_PER_WAVE; ++t) {
        const int    tileRow = rowBase + (t * WAVES_PER_WG + wave) * 16;
        const size_t qvBase  = (size_t)b * LK + tileRow;

        // Prefetch next tile's streaming rows (global_prefetch_b8)
        if (t + 1 < TILES_PER_WAVE) {
            const size_t nb = (size_t)b * LK + rowBase + ((t + 1) * WAVES_PER_WG + wave) * 16;
            __builtin_prefetch(q + (nb + m) * D_, 0, 0);
            __builtin_prefetch(v + (nb + m) * D_, 0, 0);
        }

        // ---- A fragments: q rows, fp32 -> bf16, 16-bit A layout (16x32 per kb)
        v16bf A[4];
        {
            const float* qr = q + (qvBase + m) * D_;
            #pragma unroll
            for (int kb = 0; kb < 4; ++kb) {
                const int base = kb * 32 + hi * 16;
                const float4 x0 = *(const float4*)(qr + base + 0);
                const float4 x1 = *(const float4*)(qr + base + 4);
                const float4 x2 = *(const float4*)(qr + base + 8);
                const float4 x3 = *(const float4*)(qr + base + 12);
                v16bf a;
                a[0]  = (__bf16)x0.x; a[1]  = (__bf16)x0.y;
                a[2]  = (__bf16)x0.z; a[3]  = (__bf16)x0.w;
                a[4]  = (__bf16)x1.x; a[5]  = (__bf16)x1.y;
                a[6]  = (__bf16)x1.z; a[7]  = (__bf16)x1.w;
                a[8]  = (__bf16)x2.x; a[9]  = (__bf16)x2.y;
                a[10] = (__bf16)x2.z; a[11] = (__bf16)x2.w;
                a[12] = (__bf16)x3.x; a[13] = (__bf16)x3.y;
                a[14] = (__bf16)x3.z; a[15] = (__bf16)x3.w;
                A[kb] = a;
            }
        }

        // Per-child partial row-dot accumulators
        float lacc0[8] = {0.f, 0.f, 0.f, 0.f, 0.f, 0.f, 0.f, 0.f};
        float lacc1[8] = {0.f, 0.f, 0.f, 0.f, 0.f, 0.f, 0.f, 0.f};

        // v column base for this lane: v[row = r + 8*hi][n = nt*16 + m]
        const float* vcol = v + (qvBase + 8 * hi) * D_ + m;

        // Keep nt as a REAL loop: fragment addresses depend on nt, so the
        // ds_load_b128s cannot be hoisted out of the t loop (avoids the
        // 512-VGPR LICM spill seen when everything was unrolled).
        #pragma unroll 1
        for (int nt = 0; nt < 8; ++nt) {
            float vv[8];
            #pragma unroll
            for (int r = 0; r < 8; ++r)
                vv[r] = vcol[(size_t)r * D_ + nt * 16];

            v8f acc0 = {0.f, 0.f, 0.f, 0.f, 0.f, 0.f, 0.f, 0.f};
            v8f acc1 = {0.f, 0.f, 0.f, 0.f, 0.f, 0.f, 0.f, 0.f};
            const v16bf* B0 = &Bsm[(      nt * 4) * 32 + lane];  // c = 0
            const v16bf* B1 = &Bsm[((8 + nt) * 4) * 32 + lane];  // c = 1

            #pragma unroll
            for (int kb = 0; kb < 4; ++kb) {
                // Two independent WMMA chains (c=0 / c=1) interleave to hide
                // the WMMA data hazard latency.
                acc0 = __builtin_amdgcn_wmma_f32_16x16x32_bf16(
                           false, A[kb], false, B0[kb * 32], (short)0, acc0, false, false);
                acc1 = __builtin_amdgcn_wmma_f32_16x16x32_bf16(
                           false, A[kb], false, B1[kb * 32], (short)0, acc1, false, false);
            }
            // acc VGPR r = contextual[row = r + 8*hi][n = nt*16 + m]
            #pragma unroll
            for (int r = 0; r < 8; ++r) {
                lacc0[r] += acc0[r] * vv[r];
                lacc1[r] += acc1[r] * vv[r];
            }
        }

        // Reduce over the 16 lanes of each half (columns n): xor bits 0..3,
        // then lane m==0 of each half writes both children as one float2.
        #pragma unroll
        for (int r = 0; r < 8; ++r) {
            float x0 = lacc0[r], x1 = lacc1[r];
            x0 += __shfl_xor(x0, 1); x1 += __shfl_xor(x1, 1);
            x0 += __shfl_xor(x0, 2); x1 += __shfl_xor(x1, 2);
            x0 += __shfl_xor(x0, 4); x1 += __shfl_xor(x1, 4);
            x0 += __shfl_xor(x0, 8); x1 += __shfl_xor(x1, 8);
            lacc0[r] = x0; lacc1[r] = x1;
        }
        if (m == 0) {
            #pragma unroll
            for (int r = 0; r < 8; ++r) {
                float2 o = make_float2(lacc0[r], lacc1[r]);
                *(float2*)(out + (qvBase + r + 8 * hi) * NC) = o;   // 8B store
            }
        }
    }
}

extern "C" void kernel_launch(void* const* d_in, const int* in_sizes, int n_in,
                              void* d_out, int out_size, void* d_ws, size_t ws_size,
                              hipStream_t stream) {
    (void)in_sizes; (void)n_in; (void)d_ws; (void)ws_size; (void)out_size;
    const float* q   = (const float*)d_in[0];
    const float* v   = (const float*)d_in[1];
    const float* mem = (const float*)d_in[2];
    float* out = (float*)d_out;

    dim3 grid(B_ * (LK / ROWS_PER_WG));   // 512 blocks
    dim3 block(256);                      // 8 waves (wave32)
    bmt_wmma_kernel<<<grid, block, 0, stream>>>(q, v, mem, out);
}
